// SwitchHeadAttention_32925219291619
// MI455X (gfx1250) — compile-verified, matrix operand order checked by reading, not run on hardware
//
#include <hip/hip_runtime.h>
#include <hip/hip_bf16.h>
#include <math.h>

#define DEV __device__ __forceinline__

typedef __attribute__((ext_vector_type(16))) __bf16 v16bf;
typedef __attribute__((ext_vector_type(8)))  __bf16 v8bf;
typedef __attribute__((ext_vector_type(8)))  float  v8f;

// Problem dims (compile-time)
constexpr int Bc   = 2;
constexpr int Tc   = 2048;
constexpr int DIMc = 1024;
constexpr int Hc   = 16;
constexpr int DHc  = 64;
constexpr int Ec   = 8;
constexpr int Mrows = Bc * Tc;        // 4096 tokens

// ---------------------------------------------------------------------------
// WMMA helpers (CDNA5 wave32, v_wmma_f32_16x16x32_bf16)
// ---------------------------------------------------------------------------
DEV v8f wmma_bf16(v16bf a, v16bf b, v8f c) {
  return __builtin_amdgcn_wmma_f32_16x16x32_bf16(
      /*neg_a=*/false, a, /*neg_b=*/false, b,
      /*c_mod=*/(short)0, c, /*reuse_a=*/false, /*reuse_b=*/false);
}

// Load a 16x32 A-fragment (or 32x16 B-fragment gathered as rows of Bt) from a
// row-major bf16 matrix (global or LDS). ISA layout (CDNA5 ISA §7.12.2):
//   lane<16 : row=lane,    K = {c..c+7, c+16..c+23}
//   lane>=16: row=lane-16, K = {c+8..c+15, c+24..c+31}
DEV v16bf load_frag(const __bf16* base, int stride, int lane) {
  const int r  = lane & 15;
  const int ks = (lane >> 4) & 1;
  const __bf16* p = base + (size_t)r * stride + ks * 8;
  v8bf lo = *(const v8bf*)(p);
  v8bf hi = *(const v8bf*)(p + 16);
  v16bf f;
#pragma unroll
  for (int i = 0; i < 8; ++i) { f[i] = lo[i]; f[i + 8] = hi[i]; }
  return f;
}

// ---------------------------------------------------------------------------
// K0: f32 -> bf16 convert (grid-stride)
// ---------------------------------------------------------------------------
__global__ void k_f32_to_bf16(const float* __restrict__ src,
                              __bf16* __restrict__ dst, int n) {
  int i = blockIdx.x * blockDim.x + threadIdx.x;
  int stride = gridDim.x * blockDim.x;
  for (; i < n; i += stride) dst[i] = (__bf16)src[i];
}

// ---------------------------------------------------------------------------
// K1: Y[m,n] = sum_c X[m,c] * W[n,c]      X:[4096,1024] bf16, W:[N,1024] bf16
//   mode 0: bf16 row-major [M,N]          (proj_v)
//   mode 1: bf16 scattered to [B,H,T,DH]  (q,k)
//   mode 2: f32 row-major [M,N]           (gates)
// One wave per 16x16 tile, K-loop of 32 -> 32 WMMAs/tile.
// ---------------------------------------------------------------------------
__global__ void k_gemm_bf16(const __bf16* __restrict__ X,
                            const __bf16* __restrict__ W,
                            void* __restrict__ Y, int N, int mode) {
  const int lane = threadIdx.x & 31;
  const int gw   = blockIdx.x * (blockDim.x >> 5) + (threadIdx.x >> 5);
  const int ntiles = N >> 4;
  const int tm = gw / ntiles, tn = gw % ntiles;
  if (tm >= (Mrows >> 4)) return;
  const int m0 = tm * 16, n0 = tn * 16;

  v8f acc = {};
  const __bf16* xa = X + (size_t)m0 * DIMc;
  const __bf16* wb = W + (size_t)n0 * DIMc;
#pragma unroll 4
  for (int c = 0; c < DIMc; c += 32) {
    __builtin_prefetch(xa + c + 256, 0, 0);   // global_prefetch_b8
    v16bf a = load_frag(xa + c, DIMc, lane);
    v16bf b = load_frag(wb + c, DIMc, lane);
    acc = wmma_bf16(a, b, acc);
  }

  // D layout: lane -> column n0+(lane&15); reg v -> row m0 + v + 8*(lane>>4)
  const int n  = n0 + (lane & 15);
  const int mo = (lane >> 4) * 8;
  if (mode == 0) {
    __bf16* y = (__bf16*)Y;
#pragma unroll
    for (int v = 0; v < 8; ++v) y[(size_t)(m0 + mo + v) * N + n] = (__bf16)acc[v];
  } else if (mode == 1) {
    __bf16* y = (__bf16*)Y;
    const int h = n / DHc, d = n % DHc;
#pragma unroll
    for (int v = 0; v < 8; ++v) {
      int m = m0 + mo + v;
      int bb = m / Tc, t = m % Tc;
      y[(((size_t)bb * Hc + h) * Tc + t) * DHc + d] = (__bf16)acc[v];
    }
  } else {
    float* y = (float*)Y;
#pragma unroll
    for (int v = 0; v < 8; ++v) y[(size_t)(m0 + mo + v) * N + n] = acc[v];
  }
}

// ---------------------------------------------------------------------------
// K2: V-expert top-2 sigmoid mix.  v[b,h,t,d] = sum_k sig(g_k)*proj_v[bt,e_k,d]
// Written TRANSPOSED as vT[b,h,d,t] so flash-attn V^T fragments are contiguous.
// ---------------------------------------------------------------------------
__global__ void k_moe_v(const float* __restrict__ gv,
                        const __bf16* __restrict__ pv,
                        __bf16* __restrict__ vT) {
  int idx = blockIdx.x * blockDim.x + threadIdx.x;   // (bt, h, d)
  if (idx >= Mrows * Hc * DHc) return;
  const int d  = idx % DHc;
  const int h  = (idx / DHc) % Hc;
  const int bt = idx / (DHc * Hc);
  const float* g = gv + ((size_t)bt * Hc + h) * Ec;
  float g0 = -1e30f; int e0 = 0;
#pragma unroll
  for (int e = 0; e < Ec; ++e) { float x = g[e]; if (x > g0) { g0 = x; e0 = e; } }
  float g1 = -1e30f; int e1 = 0;
#pragma unroll
  for (int e = 0; e < Ec; ++e) { float x = g[e]; if (e != e0 && x > g1) { g1 = x; e1 = e; } }
  const float w0 = 1.f / (1.f + __expf(-g0));
  const float w1 = 1.f / (1.f + __expf(-g1));
  const float val = w0 * (float)pv[((size_t)bt * Ec + e0) * DHc + d] +
                    w1 * (float)pv[((size_t)bt * Ec + e1) * DHc + d];
  const int b = bt / Tc, t = bt % Tc;
  vT[(((size_t)b * Hc + h) * DHc + d) * Tc + t] = (__bf16)val;
}

// ---------------------------------------------------------------------------
// K3: flash attention with async-to-LDS K/V staging (CDNA5 ASYNCcnt path).
// Block = 4 waves, all on the SAME (b,h); 4 consecutive 16-query tiles.
// K/V 32-key tiles are DMA'd into LDS once per block (double-buffered) via
// GLOBAL_LOAD_ASYNC_TO_LDS_B128 and shared by all 4 waves -> 4x less K/V
// HBM traffic, latency hidden behind the 8 WMMAs of the previous tile.
//
// Math: S^T = K·Q^T (queries land per-lane -> per-lane softmax stats), then
// O^T = V^T·P^T. D-layout of S^T packs directly into the B-operand of the PV
// WMMA: zero cross-lane movement.
// ---------------------------------------------------------------------------
__global__ void __launch_bounds__(128)
k_flash_attn(const __bf16* __restrict__ qb,
             const __bf16* __restrict__ kb,
             const __bf16* __restrict__ vT,
             __bf16* __restrict__ attn) {
  __shared__ alignas(128) __bf16 kbuf[2][32 * DHc];   // [buf][key][dh]  2x4KB
  __shared__ alignas(128) __bf16 vbuf[2][DHc * 32];   // [buf][dh][key]  2x4KB

  const int tid  = threadIdx.x;
  const int lane = tid & 31;
  const int wid  = tid >> 5;
  const int bh = blockIdx.x / (Tc / 64);
  const int i0 = (blockIdx.x % (Tc / 64)) * 64 + wid * 16;

  const __bf16* qp = qb + (size_t)bh * Tc * DHc;
  const __bf16* kp = kb + (size_t)bh * Tc * DHc;
  const __bf16* vp = vT + (size_t)bh * DHc * Tc;
  const float scale = 0.125f;   // DH^-0.5

  // Q as B-operand (lane = query), two dh chunks
  const v16bf qf0 = load_frag(qp + (size_t)i0 * DHc + 0,  DHc, lane);
  const v16bf qf1 = load_frag(qp + (size_t)i0 * DHc + 32, DHc, lane);

  // Issue one 32-key K/V tile into LDS buffer `buf` (4 async b128 per thread).
  auto issue = [&](int buf, int j0) {
    // K tile: rows j0..j0+31 of [T,DH] are a contiguous 4KB block.
    {
      const char* g = (const char*)(kp + (size_t)j0 * DHc) + tid * 16;
      unsigned    l = (unsigned)(size_t)(&kbuf[buf][0]) + tid * 16;
      asm volatile("global_load_async_to_lds_b128 %0, %1, off"
                   :: "v"(l), "v"(g) : "memory");
      g += 2048; l += 2048;
      asm volatile("global_load_async_to_lds_b128 %0, %1, off"
                   :: "v"(l), "v"(g) : "memory");
    }
    // V tile: 64 dh-rows x 64B chunks (row stride T*2 bytes in global).
    {
      const char* vg = (const char*)vp + (size_t)j0 * 2;
      unsigned    vb = (unsigned)(size_t)(&vbuf[buf][0]);
      int c = tid;                                   // chunk id 0..255
      const char* g = vg + (size_t)(c >> 2) * (Tc * 2) + (c & 3) * 16;
      unsigned    l = vb + c * 16;
      asm volatile("global_load_async_to_lds_b128 %0, %1, off"
                   :: "v"(l), "v"(g) : "memory");
      c += 128;
      g = vg + (size_t)(c >> 2) * (Tc * 2) + (c & 3) * 16;
      l = vb + c * 16;
      asm volatile("global_load_async_to_lds_b128 %0, %1, off"
                   :: "v"(l), "v"(g) : "memory");
    }
  };

  v8f o0 = {}, o1 = {}, o2 = {}, o3 = {};       // O^T: 64 dh x 16 queries
  float mOld = -1e30f, l = 0.f;

  constexpr int NB = Tc / 32;   // 64 key blocks
  issue(0, 0);
  for (int jb = 0; jb < NB; ++jb) {
    const int buf = jb & 1;
    if (jb + 1 < NB) {
      issue((jb + 1) & 1, (jb + 1) * 32);
      asm volatile("s_wait_asynccnt 0x4" ::: "memory");  // batch jb complete
    } else {
      asm volatile("s_wait_asynccnt 0x0" ::: "memory");
    }
    __syncthreads();   // LDS tile visible to all 4 waves

    const __bf16* kt = &kbuf[buf][0];
    const __bf16* vt = &vbuf[buf][0];

    // ---- S^T for 32 keys (two 16-key D tiles), K-dim = DH = 2x32 ----
    v8f s0 = {}, s1 = {};
    v16bf ka = load_frag(kt + 0 * DHc + 0, DHc, lane);
    s0 = wmma_bf16(ka, qf0, s0);
    ka = load_frag(kt + 0 * DHc + 32, DHc, lane);
    s0 = wmma_bf16(ka, qf1, s0);
    ka = load_frag(kt + 16 * DHc + 0, DHc, lane);
    s1 = wmma_bf16(ka, qf0, s1);
    ka = load_frag(kt + 16 * DHc + 32, DHc, lane);
    s1 = wmma_bf16(ka, qf1, s1);

    // ---- online softmax: per-query == per-lane (pairs combined via shfl16)
    float mb = -1e30f;
#pragma unroll
    for (int i = 0; i < 8; ++i) {
      s0[i] *= scale; s1[i] *= scale;
      mb = fmaxf(mb, fmaxf(s0[i], s1[i]));
    }
    mb = fmaxf(mb, __shfl_xor(mb, 16, 32));
    const float mNew = fmaxf(mOld, mb);
    const float corr = __expf(mOld - mNew);
    float lAdd = 0.f;
    v16bf pb;                       // P^T as B-operand: keys = K-dim
#pragma unroll
    for (int i = 0; i < 8; ++i) {
      float p0 = __expf(s0[i] - mNew);
      float p1 = __expf(s1[i] - mNew);
      lAdd += p0 + p1;
      pb[i] = (__bf16)p0; pb[i + 8] = (__bf16)p1;
    }
    lAdd += __shfl_xor(lAdd, 16, 32);
    l = l * corr + lAdd;
    mOld = mNew;
#pragma unroll
    for (int i = 0; i < 8; ++i) { o0[i] *= corr; o1[i] *= corr; o2[i] *= corr; o3[i] *= corr; }

    // ---- O^T += V^T(16dh x 32keys) x P^T(32keys x 16q), 4 dh chunks ----
    v16bf va = load_frag(vt + 0 * 32, 32, lane);
    o0 = wmma_bf16(va, pb, o0);
    va = load_frag(vt + 16 * 32, 32, lane);
    o1 = wmma_bf16(va, pb, o1);
    va = load_frag(vt + 32 * 32, 32, lane);
    o2 = wmma_bf16(va, pb, o2);
    va = load_frag(vt + 48 * 32, 32, lane);
    o3 = wmma_bf16(va, pb, o3);

    __syncthreads();   // all waves done with `buf` before it is overwritten
  }

  const float inv = 1.f / l;
  const int b = bh / Hc, h = bh % Hc;
  const int t  = i0 + (lane & 15);
  const int d8 = (lane >> 4) * 8;   // which half of the 16-row dh chunk
  __bf16* ap = attn + (((size_t)b * Tc + t) * Hc + h) * DHc;
  v8bf st;
#pragma unroll
  for (int i = 0; i < 8; ++i) st[i] = (__bf16)(o0[i] * inv);
  *(v8bf*)(ap + 0 + d8) = st;
#pragma unroll
  for (int i = 0; i < 8; ++i) st[i] = (__bf16)(o1[i] * inv);
  *(v8bf*)(ap + 16 + d8) = st;
#pragma unroll
  for (int i = 0; i < 8; ++i) st[i] = (__bf16)(o2[i] * inv);
  *(v8bf*)(ap + 32 + d8) = st;
#pragma unroll
  for (int i = 0; i < 8; ++i) st[i] = (__bf16)(o3[i] * inv);
  *(v8bf*)(ap + 48 + d8) = st;
}

// ---------------------------------------------------------------------------
// K4a: O-gate top-2 bitmask per (bt,h)
// ---------------------------------------------------------------------------
__global__ void k_mask_o(const float* __restrict__ go,
                         unsigned char* __restrict__ mask) {
  int idx = blockIdx.x * blockDim.x + threadIdx.x;
  if (idx >= Mrows * Hc) return;
  const float* g = go + (size_t)idx * Ec;
  float g0 = -1e30f; int e0 = 0;
#pragma unroll
  for (int e = 0; e < Ec; ++e) { float x = g[e]; if (x > g0) { g0 = x; e0 = e; } }
  float g1 = -1e30f; int e1 = 0;
#pragma unroll
  for (int e = 0; e < Ec; ++e) { float x = g[e]; if (e != e0 && x > g1) { g1 = x; e1 = e; } }
  mask[idx] = (unsigned char)((1u << e0) | (1u << e1));
}

// ---------------------------------------------------------------------------
// K4b: per-expert aggregation  u[bt,e,d] = sum_h mask[bt,h,e] * attn[bt,h,d]
// (collapses H*K=32 token-head GEMVs into E=8 -> 4x less final-GEMM work)
// ---------------------------------------------------------------------------
__global__ void k_expert_agg(const __bf16* __restrict__ attn,
                             const unsigned char* __restrict__ mask,
                             __bf16* __restrict__ u) {
  int idx = blockIdx.x * blockDim.x + threadIdx.x;  // (bt, e, d)
  if (idx >= Mrows * Ec * DHc) return;
  const int d  = idx % DHc;
  const int e  = (idx / DHc) % Ec;
  const int bt = idx / (DHc * Ec);
  float acc = 0.f;
  const __bf16* ap = attn + (size_t)bt * Hc * DHc + d;
  const unsigned char* mp = mask + (size_t)bt * Hc;
#pragma unroll
  for (int h = 0; h < Hc; ++h)
    if (mp[h] & (1u << e)) acc += (float)ap[(size_t)h * DHc];
  u[idx] = (__bf16)acc;
}

// ---------------------------------------------------------------------------
// K5: out[m,f] = sum_e sum_d u[m,e,d] * Wo[e,f,d]   (f32 out, 16 WMMAs/tile)
// ---------------------------------------------------------------------------
__global__ void k_out_gemm(const __bf16* __restrict__ u,
                           const __bf16* __restrict__ Wo,
                           float* __restrict__ out) {
  const int lane = threadIdx.x & 31;
  const int gw   = blockIdx.x * (blockDim.x >> 5) + (threadIdx.x >> 5);
  const int ntiles = DIMc >> 4;            // 64
  const int tm = gw / ntiles, tn = gw % ntiles;
  if (tm >= (Mrows >> 4)) return;
  const int m0 = tm * 16, f0 = tn * 16;

  v8f acc = {};
#pragma unroll
  for (int e = 0; e < Ec; ++e) {
    const __bf16* ua = u  + (size_t)m0 * (Ec * DHc) + e * DHc;
    const __bf16* wb = Wo + ((size_t)e * DIMc + f0) * DHc;
#pragma unroll
    for (int c = 0; c < DHc; c += 32) {
      v16bf a = load_frag(ua + c, Ec * DHc, lane);
      v16bf b = load_frag(wb + c, DHc, lane);
      acc = wmma_bf16(a, b, acc);
    }
  }
  const int f  = f0 + (lane & 15);
  const int mo = (lane >> 4) * 8;
#pragma unroll
  for (int v = 0; v < 8; ++v)
    out[(size_t)(m0 + mo + v) * DIMc + f] = acc[v];
}

// ---------------------------------------------------------------------------
// Host launcher
// ---------------------------------------------------------------------------
extern "C" void kernel_launch(void* const* d_in, const int* in_sizes, int n_in,
                              void* d_out, int out_size, void* d_ws, size_t ws_size,
                              hipStream_t stream) {
  (void)in_sizes; (void)n_in; (void)out_size; (void)ws_size;
  const float* x  = (const float*)d_in[0];
  const float* Wq = (const float*)d_in[1];
  const float* Wk = (const float*)d_in[2];
  const float* Wv = (const float*)d_in[3];
  const float* Ws = (const float*)d_in[4];
  const float* Wd = (const float*)d_in[5];
  const float* Wo = (const float*)d_in[6];
  float* out = (float*)d_out;

  // workspace carve-out (256B aligned)
  char* w = (char*)d_ws;
  auto carve = [&](size_t bytes) -> char* {
    char* p = w; w += (bytes + 255) & ~(size_t)255; return p;
  };
  __bf16* xb  = (__bf16*)carve((size_t)Mrows * DIMc * 2);          // 8 MB
  __bf16* Wqb = (__bf16*)carve((size_t)Hc * DHc * DIMc * 2);       // 2 MB
  __bf16* Wkb = (__bf16*)carve((size_t)Hc * DHc * DIMc * 2);       // 2 MB
  __bf16* Wvb = (__bf16*)carve((size_t)Ec * DHc * DIMc * 2);       // 1 MB
  __bf16* Wsb = (__bf16*)carve((size_t)Hc * Ec * DIMc * 2);        // .25 MB
  __bf16* Wdb = (__bf16*)carve((size_t)Hc * Ec * DIMc * 2);        // .25 MB
  __bf16* Wob = (__bf16*)carve((size_t)Ec * DIMc * DHc * 2);       // 1 MB
  __bf16* qbf = (__bf16*)carve((size_t)Bc * Hc * Tc * DHc * 2);    // 8 MB
  __bf16* kbf = (__bf16*)carve((size_t)Bc * Hc * Tc * DHc * 2);    // 8 MB
  __bf16* vTb = (__bf16*)carve((size_t)Bc * Hc * DHc * Tc * 2);    // 8 MB
  __bf16* pvb = (__bf16*)carve((size_t)Mrows * Ec * DHc * 2);      // 4 MB
  float*  gvb = (float*)carve((size_t)Mrows * Hc * Ec * 4);        // 2 MB
  float*  gob = (float*)carve((size_t)Mrows * Hc * Ec * 4);        // 2 MB
  __bf16* atb = (__bf16*)carve((size_t)Mrows * Hc * DHc * 2);      // 8 MB
  __bf16* ub  = (__bf16*)carve((size_t)Mrows * Ec * DHc * 2);      // 4 MB
  unsigned char* mskb = (unsigned char*)carve((size_t)Mrows * Hc); // 64 KB

  // --- K0: bf16 conversions ---
  auto cvt = [&](const float* s, __bf16* d, int n) {
    int blocks = (n + 255) / 256; if (blocks > 8192) blocks = 8192;
    hipLaunchKernelGGL(k_f32_to_bf16, dim3(blocks), dim3(256), 0, stream, s, d, n);
  };
  cvt(x,  xb,  Mrows * DIMc);
  cvt(Wq, Wqb, Hc * DHc * DIMc);
  cvt(Wk, Wkb, Hc * DHc * DIMc);
  cvt(Wv, Wvb, Ec * DHc * DIMc);
  cvt(Ws, Wsb, Hc * Ec * DIMc);
  cvt(Wd, Wdb, Hc * Ec * DIMc);
  cvt(Wo, Wob, Ec * DIMc * DHc);

  // --- K1: projections (one wave / 16x16 tile, 4 waves / block) ---
  auto gemm = [&](const __bf16* W_, void* Y, int N, int mode) {
    int waves = (Mrows / 16) * (N / 16);
    hipLaunchKernelGGL(k_gemm_bf16, dim3(waves / 4), dim3(128), 0, stream,
                       xb, W_, Y, N, mode);
  };
  gemm(Wqb, qbf, Hc * DHc, 1);   // q  -> [B,H,T,DH]
  gemm(Wkb, kbf, Hc * DHc, 1);   // k  -> [B,H,T,DH]
  gemm(Wvb, pvb, Ec * DHc, 0);   // proj_v -> [BT, E*DH] bf16
  gemm(Wsb, gvb, Hc * Ec, 2);    // gate_v -> [BT, H*E] f32
  gemm(Wdb, gob, Hc * Ec, 2);    // gate_o -> [BT, H*E] f32

  // --- K2: V expert mix -> vT[b,h,d,t] ---
  {
    int n = Mrows * Hc * DHc;
    hipLaunchKernelGGL(k_moe_v, dim3((n + 255) / 256), dim3(256), 0, stream,
                       gvb, pvb, vTb);
  }

  // --- K3: flash attention (async-to-LDS staged, double-buffered) ---
  {
    int blocks = Bc * Hc * (Tc / 64);   // 1024 blocks of 4 waves
    hipLaunchKernelGGL(k_flash_attn, dim3(blocks), dim3(128), 0, stream,
                       qbf, kbf, vTb, atb);
  }

  // --- K4: O-gate mask + per-expert aggregation ---
  {
    int n = Mrows * Hc;
    hipLaunchKernelGGL(k_mask_o, dim3((n + 255) / 256), dim3(256), 0, stream,
                       gob, mskb);
    int n2 = Mrows * Ec * DHc;
    hipLaunchKernelGGL(k_expert_agg, dim3((n2 + 255) / 256), dim3(256), 0, stream,
                       atb, mskb, ub);
  }

  // --- K5: output GEMM (fully overwrites d_out) ---
  {
    int waves = (Mrows / 16) * (DIMc / 16);  // 16384
    hipLaunchKernelGGL(k_out_gemm, dim3(waves / 4), dim3(128), 0, stream,
                       ub, Wob, out);
  }
}
